// MultiHeadAttention_88338887344707
// MI455X (gfx1250) — matrix operand order repaired and checked by reading, not compiled, not run on hardware
//
#include <hip/hip_runtime.h>
#include <hip/hip_bf16.h>
#include <math.h>

typedef _Float16 f16;
typedef __attribute__((ext_vector_type(16))) _Float16 v16h;
typedef __attribute__((ext_vector_type(8)))  _Float16 v8h;
typedef __attribute__((ext_vector_type(8)))  float    v8f;
typedef __attribute__((ext_vector_type(4)))  float    v4f;

#define DMODEL 1024
#define SLEN   2048
#define BATCH  2
#define HEADS  16
#define DK     64
#define KB     64           // key block for flash attention
#define LDS_LD (KB + 8)     // padded LDS row stride (halves)

// ---------------- WMMA helpers (CDNA5 gfx1250, wave32) ----------------

__device__ __forceinline__ v8f wmma16x16x32(v16h a, v16h b, v8f c) {
  // 8 args: (neg_a, A, neg_b, B, c_mod, C, reuse_a, reuse_b)
  return __builtin_amdgcn_wmma_f32_16x16x32_f16(false, a, false, b, (short)0, c,
                                                false, false);
}

__device__ __forceinline__ v16h combine16(v8h lo, v8h hi) {
  v16h r;
#pragma unroll
  for (int i = 0; i < 8; i++) { r[i] = lo[i]; r[i + 8] = hi[i]; }
  return r;
}

// A-matrix 16x32 f16 fragment from row-major f16 [16 x ld], element (m,k)=base[m*ld+k]
// Layout: lane m = L&15, g = L>>4; halves 0..7 -> k = 8g+i ; halves 8..15 -> k = 16+8g+i
__device__ __forceinline__ v16h load_a_f16(const f16* base, int ld, int lane) {
  int m = lane & 15, g = lane >> 4;
  const f16* r = base + m * ld + 8 * g;
  return combine16(*(const v8h*)r, *(const v8h*)(r + 16));
}

// B-matrix 32x16 f16 fragment, element (k,n) = base[n*ld + k]
// Layout: lane n = L&15, g = L>>4; halves 0..7 -> k = 8g+i ; halves 8..15 -> k = 16+8g+i
__device__ __forceinline__ v16h load_b_f16(const f16* base, int ld, int lane) {
  int n = lane & 15, g = lane >> 4;
  const f16* r = base + n * ld + 8 * g;
  return combine16(*(const v8h*)r, *(const v8h*)(r + 16));
}

// ---------------- Kernel 0: f32 -> f16 conversion (one pass) ----------------
__global__ __launch_bounds__(256) void cvt_kernel(const float* __restrict__ src,
                                                  f16* __restrict__ dst) {
  size_t i = ((size_t)blockIdx.x * 256 + threadIdx.x) * 8;
  v4f a = *(const v4f*)(src + i);
  v4f b = *(const v4f*)(src + i + 4);
  v8h o;
#pragma unroll
  for (int j = 0; j < 4; j++) { o[j] = (f16)a[j]; o[j + 4] = (f16)b[j]; }
  *(v8h*)(dst + i) = o;
}

// ---------------- Kernel 1: fused QKV projection (all f16 operands) ----------
// y = X @ W^T + b ; writes Q,K in [B,H,S,DK] f16, V transposed [B,H,DK,S] f16.
// Each wave: 32x64 output tile -> 8 WMMAs + 12 b128 loads per K-step of 32.
__global__ __launch_bounds__(256) void qkv_proj_kernel(
    const f16* __restrict__ Xq, const f16* __restrict__ Xk,
    const f16* __restrict__ Xv,
    const f16* __restrict__ Wq, const float* __restrict__ bq,
    const f16* __restrict__ Wk, const float* __restrict__ bk,
    const f16* __restrict__ Wv, const float* __restrict__ bv,
    f16* __restrict__ Qh, f16* __restrict__ Kh, f16* __restrict__ Vt) {
  int lane = threadIdx.x & 31;
  int wid  = threadIdx.x >> 5;
  int tile = blockIdx.x * 8 + wid;  // 0..2047
  int rowTile = tile >> 4;          // 0..127 (32 rows each)
  int h = tile & 15;                // column group of 64 == head
  int which = blockIdx.y;           // 0=Q 1=K 2=V

  const f16* X      = which == 0 ? Xq : (which == 1 ? Xk : Xv);
  const f16* W      = which == 0 ? Wq : (which == 1 ? Wk : Wv);
  const float* bias = which == 0 ? bq : (which == 1 ? bk : bv);

  int rbase = rowTile * 32;  // global row = b*S + s
  v8f acc0[4] = {}, acc1[4] = {};

  for (int k0 = 0; k0 < DMODEL; k0 += 32) {
    v16h a0 = load_a_f16(X + (size_t)rbase * DMODEL + k0, DMODEL, lane);
    v16h a1 = load_a_f16(X + (size_t)(rbase + 16) * DMODEL + k0, DMODEL, lane);
#pragma unroll
    for (int t = 0; t < 4; t++) {
      v16h bf = load_b_f16(W + (size_t)(h * 64 + t * 16) * DMODEL + k0, DMODEL, lane);
      acc0[t] = wmma16x16x32(a0, bf, acc0[t]);
      acc1[t] = wmma16x16x32(a1, bf, acc1[t]);
    }
  }

  int b = rbase / SLEN, s0 = rbase % SLEN;
  int g = lane >> 4, nlo = lane & 15;
#pragma unroll
  for (int t = 0; t < 4; t++) {
    int dk = t * 16 + nlo;
    float bv_ = bias[h * 64 + dk];
#pragma unroll
    for (int v = 0; v < 8; v++) {
      int m = v + 8 * g;
      f16 val0 = (f16)(acc0[t][v] + bv_);
      f16 val1 = (f16)(acc1[t][v] + bv_);
      int sA = s0 + m, sB = s0 + 16 + m;
      if (which == 2) {
        Vt[((size_t)(b * HEADS + h) * DK + dk) * SLEN + sA] = val0;  // transposed
        Vt[((size_t)(b * HEADS + h) * DK + dk) * SLEN + sB] = val1;
      } else {
        f16* dst = (which == 0) ? Qh : Kh;
        dst[((size_t)(b * HEADS + h) * SLEN + sA) * DK + dk] = val0;
        dst[((size_t)(b * HEADS + h) * SLEN + sB) * DK + dk] = val1;
      }
    }
  }
}

// ---------------- Kernel 2: flash attention ----------------
// One wave per 16-query tile of one (b,h). Online softmax over 64-key blocks.
__global__ __launch_bounds__(128) void attn_kernel(
    const f16* __restrict__ Qh, const f16* __restrict__ Kh,
    const f16* __restrict__ Vt, const int* __restrict__ mask,
    f16* __restrict__ ctx) {
  __shared__ f16 lds[4][16][LDS_LD];

  int lane = threadIdx.x & 31;
  int wid  = threadIdx.x >> 5;
  int qt = blockIdx.x * 4 + wid;  // query tile 0..127
  int bh = blockIdx.y;            // 0..31 = b*16 + h
  int b = bh >> 4, h = bh & 15;

  const f16* Qb = Qh + (size_t)bh * SLEN * DK;
  const f16* Kb = Kh + (size_t)bh * SLEN * DK;
  const f16* Vb = Vt + (size_t)bh * DK * SLEN;
  const int* mb = mask + b * SLEN;
  int qbase = qt * 16;

  // Q fragments, reused across all key blocks
  v16h aq0 = load_a_f16(Qb + (size_t)qbase * DK + 0,  DK, lane);
  v16h aq1 = load_a_f16(Qb + (size_t)qbase * DK + 32, DK, lane);

  v8f acc[4] = {};
  float m_i[8], l_i[8];
#pragma unroll
  for (int v = 0; v < 8; v++) { m_i[v] = -INFINITY; l_i[v] = 0.0f; }

  const float scale = 0.125f;  // 1/sqrt(64)
  int g = lane >> 4, nlo = lane & 15;

  for (int kb = 0; kb < SLEN; kb += KB) {
    // ---- scores S = Q K^T  (16 x 64 block, 4 subtiles of 16 keys) ----
    v8f sc[4];
#pragma unroll
    for (int nt = 0; nt < 4; nt++) {
      v8f c = {};
      v16h b0 = load_b_f16(Kb + (size_t)(kb + nt * 16) * DK + 0,  DK, lane);
      c = wmma16x16x32(aq0, b0, c);
      v16h b1 = load_b_f16(Kb + (size_t)(kb + nt * 16) * DK + 32, DK, lane);
      c = wmma16x16x32(aq1, b1, c);
      sc[nt] = c;
    }

    // ---- mask + scale ----
#pragma unroll
    for (int nt = 0; nt < 4; nt++) {
      int mv = mb[kb + nt * 16 + nlo];
#pragma unroll
      for (int v = 0; v < 8; v++)
        sc[nt][v] = mv ? sc[nt][v] * scale : -1.0e9f;
    }

    // ---- online softmax (row stats across 16 lanes per half-wave) ----
    float alpha[8];
#pragma unroll
    for (int v = 0; v < 8; v++) {
      float mx = fmaxf(fmaxf(sc[0][v], sc[1][v]), fmaxf(sc[2][v], sc[3][v]));
#pragma unroll
      for (int off = 1; off < 16; off <<= 1)
        mx = fmaxf(mx, __shfl_xor(mx, off, 32));
      float mnew = fmaxf(m_i[v], mx);
      alpha[v] = __expf(m_i[v] - mnew);
      m_i[v] = mnew;
      float rs = 0.0f;
#pragma unroll
      for (int nt = 0; nt < 4; nt++) {
        float p = __expf(sc[nt][v] - mnew);
        sc[nt][v] = p;
        rs += p;
      }
#pragma unroll
      for (int off = 1; off < 16; off <<= 1)
        rs += __shfl_xor(rs, off, 32);
      l_i[v] = l_i[v] * alpha[v] + rs;
    }

    // ---- rescale running ctx accumulators ----
#pragma unroll
    for (int t = 0; t < 4; t++)
#pragma unroll
      for (int v = 0; v < 8; v++) acc[t][v] *= alpha[v];

    // ---- P (C layout) -> LDS -> A layout (same wave, no barrier) ----
#pragma unroll
    for (int nt = 0; nt < 4; nt++)
#pragma unroll
      for (int v = 0; v < 8; v++)
        lds[wid][v + 8 * g][nt * 16 + nlo] = (f16)sc[nt][v];
    asm volatile("s_wait_dscnt 0" ::: "memory");
    v16h pa0 = load_a_f16(&lds[wid][0][0], LDS_LD, lane);
    v16h pa1 = load_a_f16(&lds[wid][0][32], LDS_LD, lane);

    // ---- ctx += P @ V   (V transposed: (k,n) = Vb[n*SLEN + k], contiguous k) ----
#pragma unroll
    for (int t = 0; t < 4; t++) {
      v16h bv0 = load_b_f16(Vb + (size_t)(t * 16) * SLEN + kb, SLEN, lane);
      acc[t] = wmma16x16x32(pa0, bv0, acc[t]);
      v16h bv1 = load_b_f16(Vb + (size_t)(t * 16) * SLEN + kb + 32, SLEN, lane);
      acc[t] = wmma16x16x32(pa1, bv1, acc[t]);
    }
  }

  // ---- normalize and write ctx [B,S,DMODEL] f16 ----
#pragma unroll
  for (int v = 0; v < 8; v++) l_i[v] = 1.0f / l_i[v];
#pragma unroll
  for (int t = 0; t < 4; t++) {
#pragma unroll
    for (int v = 0; v < 8; v++) {
      int m = v + 8 * g;
      int s = qbase + m;
      int col = h * 64 + t * 16 + nlo;
      ctx[(size_t)(b * SLEN + s) * DMODEL + col] = (f16)(acc[t][v] * l_i[v]);
    }
  }
}

// ---------------- Kernel 3: output projection (32x64 tile / wave) ----------
__global__ __launch_bounds__(256) void out_proj_kernel(
    const f16* __restrict__ ctx, const f16* __restrict__ Wo,
    const float* __restrict__ bo, float* __restrict__ out) {
  int lane = threadIdx.x & 31;
  int wid  = threadIdx.x >> 5;
  int tile = blockIdx.x * 8 + wid;  // 0..2047
  int rowTile = tile >> 4, cg = tile & 15;
  int rbase = rowTile * 32, n0 = cg * 64;

  v8f acc0[4] = {}, acc1[4] = {};
  for (int k0 = 0; k0 < DMODEL; k0 += 32) {
    v16h a0 = load_a_f16(ctx + (size_t)rbase * DMODEL + k0, DMODEL, lane);
    v16h a1 = load_a_f16(ctx + (size_t)(rbase + 16) * DMODEL + k0, DMODEL, lane);
#pragma unroll
    for (int t = 0; t < 4; t++) {
      v16h bf = load_b_f16(Wo + (size_t)(n0 + t * 16) * DMODEL + k0, DMODEL, lane);
      acc0[t] = wmma16x16x32(a0, bf, acc0[t]);
      acc1[t] = wmma16x16x32(a1, bf, acc1[t]);
    }
  }

  int g = lane >> 4, nlo = lane & 15;
#pragma unroll
  for (int t = 0; t < 4; t++) {
    float bias = bo[n0 + t * 16 + nlo];
#pragma unroll
    for (int v = 0; v < 8; v++) {
      int m = v + 8 * g;
      out[(size_t)(rbase + m) * DMODEL + n0 + t * 16 + nlo] = acc0[t][v] + bias;
      out[(size_t)(rbase + 16 + m) * DMODEL + n0 + t * 16 + nlo] = acc1[t][v] + bias;
    }
  }
}

// ---------------- launcher ----------------
extern "C" void kernel_launch(void* const* d_in, const int* in_sizes, int n_in,
                              void* d_out, int out_size, void* d_ws, size_t ws_size,
                              hipStream_t stream) {
  const float* query = (const float*)d_in[0];
  const float* key   = (const float*)d_in[1];
  const float* value = (const float*)d_in[2];
  const int*   mask  = (const int*)d_in[3];
  const float* Wq = (const float*)d_in[4];
  const float* bq = (const float*)d_in[5];
  const float* Wk = (const float*)d_in[6];
  const float* bk = (const float*)d_in[7];
  const float* Wv = (const float*)d_in[8];
  const float* bv = (const float*)d_in[9];
  const float* Wo = (const float*)d_in[10];
  const float* bo = (const float*)d_in[11];
  float* out = (float*)d_out;

  const size_t NROW = (size_t)BATCH * SLEN;   // 4096
  const size_t XSZ  = NROW * DMODEL;          // 4 Mi elements
  const size_t WSZ  = (size_t)DMODEL * DMODEL;  // 1 Mi elements

  f16* p = (f16*)d_ws;
  f16* Xq = p;            p += XSZ;   // f16 copies of activations
  f16* Xk = p;            p += XSZ;
  f16* Xv = p;            p += XSZ;
  f16* Whq = p;           p += WSZ;   // f16 copies of weights
  f16* Whk = p;           p += WSZ;
  f16* Whv = p;           p += WSZ;
  f16* Who = p;           p += WSZ;
  f16* Qh  = p;           p += XSZ;   // [B,H,S,DK]
  f16* Kh  = p;           p += XSZ;   // [B,H,S,DK]
  f16* Vt  = p;           p += XSZ;   // [B,H,DK,S]
  f16* ctx = p;           p += XSZ;   // [B,S,DMODEL]
  (void)ws_size; (void)in_sizes; (void)n_in; (void)out_size;

  // one-time f32 -> f16 conversions (bandwidth-trivial)
  dim3 blkC(256);
  cvt_kernel<<<dim3((unsigned)(XSZ / 8 / 256)), blkC, 0, stream>>>(query, Xq);
  cvt_kernel<<<dim3((unsigned)(XSZ / 8 / 256)), blkC, 0, stream>>>(key, Xk);
  cvt_kernel<<<dim3((unsigned)(XSZ / 8 / 256)), blkC, 0, stream>>>(value, Xv);
  cvt_kernel<<<dim3((unsigned)(WSZ / 8 / 256)), blkC, 0, stream>>>(Wq, Whq);
  cvt_kernel<<<dim3((unsigned)(WSZ / 8 / 256)), blkC, 0, stream>>>(Wk, Whk);
  cvt_kernel<<<dim3((unsigned)(WSZ / 8 / 256)), blkC, 0, stream>>>(Wv, Whv);
  cvt_kernel<<<dim3((unsigned)(WSZ / 8 / 256)), blkC, 0, stream>>>(Wo, Who);

  dim3 blkP(256), grdP(256, 3);   // 2048 32x64 tiles per matrix / 8 waves
  qkv_proj_kernel<<<grdP, blkP, 0, stream>>>(Xq, Xk, Xv, Whq, bq, Whk, bk,
                                             Whv, bv, Qh, Kh, Vt);

  dim3 blkA(128), grdA(32, 32);   // 128 query tiles / 4 waves, 32 (b,h)
  attn_kernel<<<grdA, blkA, 0, stream>>>(Qh, Kh, Vt, mask, ctx);

  dim3 blkO(256), grdO(256);
  out_proj_kernel<<<grdO, blkO, 0, stream>>>(ctx, Who, bo, out);
}
